// MoEElementFusion_42262478192784
// MI455X (gfx1250) — matrix operand order repaired and checked by reading, not compiled, not run on hardware
//
#include <hip/hip_runtime.h>
#include <math.h>

// Problem constants (from reference: B=2, L=1024, D=512, E=16, F=4D=2048, top-4)
#define NE    16
#define DD    512
#define FF    2048
#define NTOK  2048          // B*L
#define TOPK  4
#define MT    64            // tokens per FFN tile
#define FC    64            // F-chunk per pipeline stage
#define XSTR  520           // padded bf16 stride for X tile in LDS (bank-conflict free)
#define HSTR  72            // padded bf16 stride for H1 tile in LDS

typedef __attribute__((ext_vector_type(16))) __bf16 v16bf;
typedef __attribute__((ext_vector_type(8)))  float  v8f;

// ---------- bf16 helpers (store bits as unsigned short; RNE convert) ----------
__device__ __forceinline__ unsigned short f2bf(float x) {
    unsigned int u = __float_as_uint(x);
    u += 0x7FFFu + ((u >> 16) & 1u);      // round-to-nearest-even
    return (unsigned short)(u >> 16);
}

__device__ __forceinline__ v8f zero8() {
    v8f z = {0.f, 0.f, 0.f, 0.f, 0.f, 0.f, 0.f, 0.f};
    return z;
}

// A-fragment (16x32 bf16, M x K) from LDS row-major [rows][stride].
// ISA 7.12.2: row m lives in lanes m and m+16; lanes 0-15 hold K 0-7 & 16-23,
// lanes 16-31 hold K 8-15 & 24-31 (2 bf16 per VGPR, sequential).
__device__ __forceinline__ v16bf load_a_lds(const unsigned short* base, int m0,
                                            int stride, int kb, int lane) {
    const unsigned short* p =
        base + (m0 + (lane & 15)) * stride + kb + ((lane >> 4) << 3);
    union { v16bf v; uint4 q[2]; } f;
    f.q[0] = *(const uint4*)p;            // K = kb + sel*8 + [0..7]
    f.q[1] = *(const uint4*)(p + 16);     // K = kb + 16 + sel*8 + [0..7]
    return f.v;
}

// B-fragment (32x16 bf16, K x N) from global, K contiguous per column row.
// col n in lanes n (K 0-7 & 16-23) and n+16 (K 8-15 & 24-31), matching A.
__device__ __forceinline__ v16bf load_b_glb(const unsigned short* p) {
    union { v16bf v; uint4 q[2]; } f;
    f.q[0] = *(const uint4*)p;
    f.q[1] = *(const uint4*)(p + 16);
    return f.v;
}

__device__ __forceinline__ v8f wmma_bf16(v16bf a, v16bf b, v8f c) {
    return __builtin_amdgcn_wmma_f32_16x16x32_bf16(
        /*neg_a=*/false, a, /*neg_b=*/false, b,
        /*c_mod=*/(short)0, c, /*reuse_a=*/false, /*reuse_b=*/false);
}

// =====================================================================
// Kernel 1: zero output + routing counters
// =====================================================================
__global__ void mf_zero_kernel(float* __restrict__ out, int n,
                               int* __restrict__ cnt, int ncnt) {
    int i = blockIdx.x * 256 + threadIdx.x;
    if (i < n)    out[i] = 0.f;
    if (i < ncnt) cnt[i] = 0;
}

// =====================================================================
// Kernel 2: per-expert transpose + fp32 -> bf16 convert
// in:  [E][R][C] fp32   out: [E][C][R] bf16   (R, C multiples of 32)
// =====================================================================
__global__ void mf_transpose_cvt(const float* __restrict__ in,
                                 unsigned short* __restrict__ out,
                                 int R, int C) {
    __shared__ float tile[32][33];
    const int e  = blockIdx.z;
    const int c0 = blockIdx.x * 32, r0 = blockIdx.y * 32;
    const int tx = threadIdx.x, ty = threadIdx.y;       // 32 x 8
    const float* src = in + (size_t)e * R * C;
    unsigned short* dst = out + (size_t)e * R * C;
#pragma unroll
    for (int i = 0; i < 32; i += 8)
        tile[ty + i][tx] = src[(size_t)(r0 + ty + i) * C + (c0 + tx)];
    __syncthreads();
#pragma unroll
    for (int i = 0; i < 32; i += 8)
        dst[(size_t)(c0 + ty + i) * R + (r0 + tx)] = f2bf(tile[tx][ty + i]);
}

// =====================================================================
// Kernel 3: Laplace gate + top-4 softmax routing (one wave32 per token)
// =====================================================================
__global__ __launch_bounds__(256) void mf_gate_kernel(
        const float* __restrict__ view0, const float* __restrict__ view1,
        const float* __restrict__ keys,                  // [E][D]
        const float* __restrict__ Wr,                    // [V][D][E]
        const float* __restrict__ br,                    // [V][E]
        int* __restrict__ cnt, int* __restrict__ tokIdx, // [V][E], [V][E][NTOK]
        float* __restrict__ tokW) {                      // [V][E][NTOK]
    const int v     = blockIdx.y;
    const int token = blockIdx.x * 8 + (threadIdx.x >> 5);
    const int lane  = threadIdx.x & 31;
    const float* h  = (v ? view1 : view0) + (size_t)token * DD;

    float hreg[16], hh = 0.f;
#pragma unroll
    for (int i = 0; i < 16; ++i) {
        hreg[i] = h[lane + i * 32];
        hh += hreg[i] * hreg[i];
    }
#pragma unroll
    for (int off = 16; off; off >>= 1) hh += __shfl_xor(hh, off, 32);

    float logits[NE];
    for (int e = 0; e < NE; ++e) {
        const float* ke = keys + e * DD;
        float dk = 0.f, kk = 0.f, rr = 0.f;
#pragma unroll
        for (int i = 0; i < 16; ++i) {
            const int d = lane + i * 32;
            const float kv = ke[d];
            dk += hreg[i] * kv;
            kk += kv * kv;
            rr += hreg[i] * Wr[((size_t)v * DD + d) * NE + e];
        }
#pragma unroll
        for (int off = 16; off; off >>= 1) {
            dk += __shfl_xor(dk, off, 32);
            kk += __shfl_xor(kk, off, 32);
            rr += __shfl_xor(rr, off, 32);
        }
        const float sq = fmaxf(hh - 2.f * dk + kk, 0.f);
        logits[e] = -sq + rr + br[v * NE + e];
    }

    if (lane == 0) {
        int   idx[TOPK];
        float val[TOPK];
        unsigned used = 0;
#pragma unroll
        for (int k = 0; k < TOPK; ++k) {            // stable top-4 (ties -> lowest idx)
            float best = -INFINITY; int bi = 0;
            for (int e = 0; e < NE; ++e)
                if (!((used >> e) & 1) && logits[e] > best) { best = logits[e]; bi = e; }
            used |= 1u << bi; idx[k] = bi; val[k] = best;
        }
        const float mx = val[0];
        float ex[TOPK], sum = 0.f;
#pragma unroll
        for (int k = 0; k < TOPK; ++k) { ex[k] = expf(val[k] - mx); sum += ex[k]; }
        const float inv = 1.f / sum;
#pragma unroll
        for (int k = 0; k < TOPK; ++k) {
            const int e   = idx[k];
            const int pos = atomicAdd(&cnt[v * NE + e], 1);
            tokIdx[((size_t)v * NE + e) * NTOK + pos] = token;
            tokW [((size_t)v * NE + e) * NTOK + pos] = ex[k] * inv;
        }
    }
}

// =====================================================================
// Kernel 4: routed expert FFN via WMMA bf16 (fp32 accumulate)
// grid: (32 token-tiles, E, V), block: 256 (8 waves)
// GEMM1: wave owns F-tile ft=w&3, M-tiles {2*(w>>2), 2*(w>>2)+1}
//        -> one loop-invariant B-load feeds 2 WMMAs (A reused from LDS)
// GEMM2: wave owns N-tiles 4w..4w+3 across all 4 M-tiles, B explicitly
//        hoisted: 8 global loads -> 16 WMMAs per k-step
// =====================================================================
__global__ __launch_bounds__(256) void mf_ffn_kernel(
        const float* __restrict__ view0, const float* __restrict__ view1,
        const unsigned short* __restrict__ W1t,   // [E][F][D] bf16
        const unsigned short* __restrict__ W2t,   // [E][D][F] bf16
        const float* __restrict__ b1,             // [E][F]
        const float* __restrict__ b2,             // [E][D]
        const int* __restrict__ cnt,
        const int* __restrict__ tokIdx,
        const float* __restrict__ tokW,
        float* __restrict__ out) {                // [NTOK][D]
    __shared__ __align__(16) unsigned short ldsX[MT * XSTR];  // 65 KB
    __shared__ __align__(16) unsigned short ldsH[MT * HSTR];  // 9 KB
    __shared__ float ldsWt[MT];
    __shared__ int   ldsTok[MT];

    const int tile = blockIdx.x, e = blockIdx.y, v = blockIdx.z;
    const int count = cnt[v * NE + e];
    const int base  = tile * MT;
    if (base >= count) return;
    const int nvalid = min(MT, count - base);

    const float* view = v ? view1 : view0;
    const int tid  = threadIdx.x;
    const int lane = tid & 31;
    const int w    = tid >> 5;            // wave 0..7
    const int lm   = lane & 15;           // N index within a tile
    const int sel  = lane >> 4;           // half-select

    // ---- stage token ids / combine weights ----
    if (tid < MT) {
        const int m = tid;
        ldsTok[m] = (m < nvalid) ? tokIdx[((size_t)v * NE + e) * NTOK + base + m] : -1;
        ldsWt[m]  = (m < nvalid) ? tokW [((size_t)v * NE + e) * NTOK + base + m] : 0.f;
    }
    __syncthreads();

    // ---- gather X tile [64][512] fp32 -> bf16 LDS (coalesced float4 reads) ----
    for (int m = w; m < MT; m += 8) {
        const int t = ldsTok[m];
        const float4* src = (const float4*)(view + (size_t)(t < 0 ? 0 : t) * DD);
#pragma unroll
        for (int k = 0; k < 4; ++k) {
            const int d4 = lane + k * 32;       // float4 index 0..127
            float4 x = (t >= 0) ? src[d4] : make_float4(0.f, 0.f, 0.f, 0.f);
            union { unsigned short s[4]; uint2 u; } p;
            p.s[0] = f2bf(x.x); p.s[1] = f2bf(x.y);
            p.s[2] = f2bf(x.z); p.s[3] = f2bf(x.w);
            *(uint2*)&ldsX[m * XSTR + d4 * 4] = p.u;
        }
    }

    // ---- persistent Y accumulators: wave w owns N-tiles 4w..4w+3, all 4 M-tiles ----
    v8f Yacc[4][4];
#pragma unroll
    for (int mt = 0; mt < 4; ++mt)
#pragma unroll
        for (int j = 0; j < 4; ++j) Yacc[mt][j] = zero8();

    const int ft  = w & 3;                // GEMM1: this wave's F-tile within chunk
    const int mth = w >> 2;               // GEMM1: handles M-tiles 2*mth, 2*mth+1

    __syncthreads();

    for (int fc = 0; fc < FF; fc += FC) {               // 32 F-chunks
        // ---------- GEMM1: H1 tiles (2*mth+q, ft) = X @ W1[e][:, fc+ft*16..] ----------
        v8f hacc[2] = { zero8(), zero8() };
#pragma unroll 4
        for (int kb = 0; kb < DD; kb += 32) {
            const unsigned short* bp =
                W1t + ((size_t)e * FF + fc + ft * 16 + lm) * DD + kb + sel * 8;
            const v16bf b = load_b_glb(bp);             // loop-invariant in q
#pragma unroll
            for (int q = 0; q < 2; ++q) {
                const v16bf a = load_a_lds(ldsX, (2 * mth + q) * 16, XSTR, kb, lane);
                hacc[q] = wmma_bf16(a, b, hacc[q]);
            }
        }
        __syncthreads();   // previous chunk's ldsH fully consumed by GEMM2

        // gelu(x) = 0.5 x (1 + erf(x/sqrt(2))), then bf16 -> ldsH
        {
            const int fl = ft * 16 + lm;                // local col in chunk
            const float bb = b1[(size_t)e * FF + fc + fl];
#pragma unroll
            for (int q = 0; q < 2; ++q) {
#pragma unroll
                for (int r = 0; r < 8; ++r) {
                    const int m = (2 * mth + q) * 16 + sel * 8 + r;
                    const float x = hacc[q][r] + bb;
                    const float g = 0.5f * x * (1.0f + erff(x * 0.70710678118654752f));
                    ldsH[m * HSTR + fl] = f2bf(g);
                }
            }
        }
        __syncthreads();

        // ---------- GEMM2: Y += H1[64,64] @ W2[e][fc..fc+63, :] ----------
#pragma unroll
        for (int kb = 0; kb < FC; kb += 32) {
            v16bf bfr[4];                               // explicit B hoist
#pragma unroll
            for (int j = 0; j < 4; ++j) {
                const int n = (w * 4 + j) * 16;         // output D column tile
                bfr[j] = load_b_glb(
                    W2t + ((size_t)e * DD + n + lm) * FF + fc + kb + sel * 8);
            }
#pragma unroll
            for (int mt = 0; mt < 4; ++mt) {
                const v16bf a = load_a_lds(ldsH, mt * 16, HSTR, kb, lane);
#pragma unroll
                for (int j = 0; j < 4; ++j)
                    Yacc[mt][j] = wmma_bf16(a, bfr[j], Yacc[mt][j]);
            }
        }
    }

    // ---- epilogue: out[t, n] += comb_w * (Y + b2) ----
#pragma unroll
    for (int mt = 0; mt < 4; ++mt) {
#pragma unroll
        for (int j = 0; j < 4; ++j) {
            const int n  = (w * 4 + j) * 16 + lm;
            const float bb = b2[(size_t)e * DD + n];
#pragma unroll
            for (int r = 0; r < 8; ++r) {
                const int m = mt * 16 + sel * 8 + r;
                const int t = ldsTok[m];
                if (t >= 0) {
                    const float y = (Yacc[mt][j][r] + bb) * ldsWt[m];
                    unsafeAtomicAdd(&out[(size_t)t * DD + n], y);
                }
            }
        }
    }
}

// =====================================================================
// Launch
// =====================================================================
extern "C" void kernel_launch(void* const* d_in, const int* in_sizes, int n_in,
                              void* d_out, int out_size, void* d_ws, size_t ws_size,
                              hipStream_t stream) {
    const float* view0 = (const float*)d_in[0];
    const float* view1 = (const float*)d_in[1];
    const float* keys  = (const float*)d_in[2];
    const float* W1    = (const float*)d_in[3];
    const float* b1    = (const float*)d_in[4];
    const float* W2    = (const float*)d_in[5];
    const float* b2    = (const float*)d_in[6];
    const float* Wr    = (const float*)d_in[7];
    const float* br    = (const float*)d_in[8];
    float* out = (float*)d_out;

    // Workspace layout (needs ~65 MB):
    //   [0,32MB)    W1t bf16 [E][F][D]
    //   [32,64MB)   W2t bf16 [E][D][F]
    //   +0          cnt    [V][E] int      (4 KB pad)
    //   +4KB        tokIdx [V][E][NTOK] int (256 KB)
    //   +4KB+256KB  tokW   [V][E][NTOK] f32 (256 KB)
    char* ws = (char*)d_ws;
    unsigned short* W1t = (unsigned short*)ws;
    unsigned short* W2t = (unsigned short*)(ws + (size_t)32 * 1024 * 1024);
    char* meta = ws + (size_t)64 * 1024 * 1024;
    int*   cnt  = (int*)meta;
    int*   tokI = (int*)(meta + 4096);
    float* tokW = (float*)(meta + 4096 + (size_t)2 * NE * NTOK * sizeof(int));

    const int nOut = NTOK * DD;   // 1,048,576
    mf_zero_kernel<<<(nOut + 255) / 256, 256, 0, stream>>>(out, nOut, cnt, 2 * NE);

    dim3 tb(32, 8);
    // W1 [E][D][F] -> W1t [E][F][D]
    mf_transpose_cvt<<<dim3(FF / 32, DD / 32, NE), tb, 0, stream>>>(W1, W1t, DD, FF);
    // W2 [E][F][D] -> W2t [E][D][F]
    mf_transpose_cvt<<<dim3(DD / 32, FF / 32, NE), tb, 0, stream>>>(W2, W2t, FF, DD);

    mf_gate_kernel<<<dim3(NTOK / 8, 2), 256, 0, stream>>>(
        view0, view1, keys, Wr, br, cnt, tokI, tokW);

    mf_ffn_kernel<<<dim3(NTOK / MT, NE, 2), 256, 0, stream>>>(
        view0, view1, W1t, W2t, b1, b2, cnt, tokI, tokW, out);
}
